// LatentSelfAttention_42193758716042
// MI455X (gfx1250) — matrix operand order repaired and checked by reading, not compiled, not run on hardware
//
#include <hip/hip_runtime.h>

// ---------------------------------------------------------------------------
// MLA latent self-attention for MI455X (gfx1250, wave32, WMMA).
// Heavy math: v_wmma_f32_16x16x32_f16 (f32 accumulate).
// Global->LDS staging: GLOBAL_LOAD_ASYNC_TO_LDS_B128 (ASYNCcnt) with
// double-buffered LDS so DMA overlaps WMMA issue.
// ---------------------------------------------------------------------------

typedef _Float16 h16;
typedef __attribute__((ext_vector_type(16))) _Float16 v16h;
typedef __attribute__((ext_vector_type(8)))  float    v8f;

union Frag { v16h v; uint4 q[2]; };

#ifndef LSA_B
#define LSA_B 4
#define LSA_S 4096
#define LSA_E 2048
#define LSA_L 512
#endif

// Async DMA: global (64-bit vaddr) -> LDS (32-bit dest offset in vdst VGPR).
__device__ __forceinline__ void lsa_async_cp16(unsigned lds_off, const void* gptr)
{
    asm volatile("global_load_async_to_lds_b128 %0, %1, off"
                 :: "v"(lds_off), "v"(gptr)
                 : "memory");
}
__device__ __forceinline__ void lsa_wait_async0()
{
    asm volatile("s_wait_asynccnt 0x0" ::: "memory");
}

// ---------------------------------------------------------------------------
// Generic fp16 GEMM:  C[M,N] = A[M,K] * Bt[N,K]^T   (both row-major over K)
// OUT_MODE: 0 = f32 row-major, 1 = fp16 row-major, 2 = fp16 transposed (C^T)
// CAUSAL_SKIP: skip tiles entirely above the diagonal (scores GEMM)
// Tiling: 256 threads = 8 waves, tile 128x128x32, double-buffered async LDS.
// Wave (wrow,wcol) in 2x4 grid computes a 64x32 patch = 4x2 WMMA tiles.
// ---------------------------------------------------------------------------
template <int OUT_MODE, bool CAUSAL_SKIP>
__global__ __launch_bounds__(256) void lsa_gemm(
    const h16* __restrict__ A, const h16* __restrict__ Bt, void* __restrict__ C,
    int M, int N, int K,
    long long batchA, long long batchB, long long batchC)
{
    constexpr int BM = 128, BN = 128, BK = 32;
    __shared__ h16 As[2][BM * BK];   // 2 x 8 KB
    __shared__ h16 Bs[2][BN * BK];   // 2 x 8 KB

    const int bz  = blockIdx.z;
    A  += (size_t)bz * (size_t)batchA;
    Bt += (size_t)bz * (size_t)batchB;

    const int bm0 = blockIdx.y * BM;
    const int bn0 = blockIdx.x * BN;
    if (CAUSAL_SKIP && bn0 > bm0 + (BM - 1)) return;   // fully-masked tile

    const int tid  = threadIdx.x;
    const int wave = tid >> 5;
    const int lane = tid & 31;
    const int wrow = wave >> 2;   // 0..1 : rows  wrow*64 .. +63
    const int wcol = wave & 3;    // 0..3 : cols  wcol*32 .. +31
    const int lhalf = lane >> 4;  // 0|1
    const int lmod  = lane & 15;

    // --- per-thread DMA assignment: 4 x 16B chunks per K-tile --------------
    // chunk index space: 512 chunks per 128x32 tile; thread t owns t and t+256
    const int r0 = tid >> 2;                 // rows 0..63
    const int r1 = r0 + 64;                  // rows 64..127
    const int cv = (tid & 3) * 8;            // h16 column offset within BK

    const h16* gA0 = A  + (size_t)(bm0 + r0) * K + cv;
    const h16* gA1 = A  + (size_t)(bm0 + r1) * K + cv;
    const h16* gB0 = Bt + (size_t)(bn0 + r0) * K + cv;
    const h16* gB1 = Bt + (size_t)(bn0 + r1) * K + cv;

    unsigned lA0[2], lA1[2], lB0[2], lB1[2];
    #pragma unroll
    for (int b2 = 0; b2 < 2; ++b2) {
        lA0[b2] = (unsigned)(size_t)&As[b2][r0 * BK + cv];
        lA1[b2] = (unsigned)(size_t)&As[b2][r1 * BK + cv];
        lB0[b2] = (unsigned)(size_t)&Bs[b2][r0 * BK + cv];
        lB1[b2] = (unsigned)(size_t)&Bs[b2][r1 * BK + cv];
    }

    auto issue_tile = [&](int kt, int buf) {
        const size_t ko = (size_t)kt * BK;
        lsa_async_cp16(lA0[buf], gA0 + ko);
        lsa_async_cp16(lA1[buf], gA1 + ko);
        lsa_async_cp16(lB0[buf], gB0 + ko);
        lsa_async_cp16(lB1[buf], gB1 + ko);
    };

    v8f acc[4][2] = {};

    const int ntile = K / BK;
    issue_tile(0, 0);
    lsa_wait_async0();
    __syncthreads();

    for (int t = 0; t < ntile; ++t) {
        const int cur = t & 1;
        if (t + 1 < ntile) issue_tile(t + 1, cur ^ 1);   // DMA overlaps WMMA

        // --- fragment loads per ISA 7.12.2 layouts -------------------------
        Frag af[4], bfg[2];
        #pragma unroll
        for (int mt = 0; mt < 4; ++mt) {
            // A 16x32: lanes 0-15 row M=l hold K 0-7/16-23; lanes 16-31 hold 8-15/24-31
            const h16* p = &As[cur][(size_t)(wrow * 64 + mt * 16 + lmod) * BK + lhalf * 8];
            af[mt].q[0] = *(const uint4*)p;
            af[mt].q[1] = *(const uint4*)(p + 16);
        }
        #pragma unroll
        for (int nt = 0; nt < 2; ++nt) {
            // B 32x16: lanes 0-15 col N=l hold K 0-7/16-23; lanes 16-31 hold 8-15/24-31
            const h16* p = &Bs[cur][(size_t)(wcol * 32 + nt * 16 + lmod) * BK + lhalf * 8];
            bfg[nt].q[0] = *(const uint4*)p;
            bfg[nt].q[1] = *(const uint4*)(p + 16);
        }

        #pragma unroll
        for (int mt = 0; mt < 4; ++mt)
            #pragma unroll
            for (int nt = 0; nt < 2; ++nt)
                acc[mt][nt] = __builtin_amdgcn_wmma_f32_16x16x32_f16(
                    false, af[mt].v, false, bfg[nt].v,
                    (short)0, acc[mt][nt], false, false);

        lsa_wait_async0();      // next tile's DMA complete (own wave)
        __syncthreads();        // all waves: reads done + new tile visible
    }

    // --- epilogue: C/D layout = VGPR r -> M = r + 8*lhalf, N = lmod ---------
    const size_t cbase = (size_t)bz * (size_t)batchC;
    #pragma unroll
    for (int mt = 0; mt < 4; ++mt) {
        #pragma unroll
        for (int nt = 0; nt < 2; ++nt) {
            const int n = bn0 + wcol * 32 + nt * 16 + lmod;
            #pragma unroll
            for (int r = 0; r < 8; ++r) {
                const int m = bm0 + wrow * 64 + mt * 16 + r + lhalf * 8;
                const float val = acc[mt][nt][r];
                if (OUT_MODE == 0)
                    ((float*)C)[cbase + (size_t)m * N + n] = val;
                else if (OUT_MODE == 1)
                    ((h16*)C)[cbase + (size_t)m * N + n] = (h16)val;
                else
                    ((h16*)C)[cbase + (size_t)n * M + m] = (h16)val;  // C^T
            }
        }
    }
}

// ---------------------------------------------------------------------------
// Elementwise f32 -> fp16 convert
// ---------------------------------------------------------------------------
__global__ void lsa_conv(const float* __restrict__ src, h16* __restrict__ dst, size_t n)
{
    size_t i = (size_t)blockIdx.x * blockDim.x + threadIdx.x;
    if (i < n) dst[i] = (h16)src[i];
}

// ---------------------------------------------------------------------------
// Transposing f32 [R,C] -> fp16 [C,R] convert (LDS-tiled 32x32)
// ---------------------------------------------------------------------------
__global__ __launch_bounds__(256) void lsa_tconv(
    const float* __restrict__ src, h16* __restrict__ dst, int R, int Ccols)
{
    __shared__ float t[32][33];
    const int c0 = blockIdx.x * 32, r0 = blockIdx.y * 32;
    const int tx = threadIdx.x & 31, ty = threadIdx.x >> 5;  // ty 0..7
    #pragma unroll
    for (int i = 0; i < 32; i += 8)
        t[ty + i][tx] = src[(size_t)(r0 + ty + i) * Ccols + c0 + tx];
    __syncthreads();
    #pragma unroll
    for (int i = 0; i < 32; i += 8)
        dst[(size_t)(c0 + ty + i) * R + r0 + tx] = (h16)t[tx][ty + i];
}

// ---------------------------------------------------------------------------
// LayerNorm over L=512 per row; one wave32 per row (16 elems/lane).
// Writes fp32 (cached c_kv output) and fp16 (for downstream WMMA GEMMs).
// ---------------------------------------------------------------------------
__global__ __launch_bounds__(256) void lsa_layernorm(
    const float* __restrict__ src, const float* __restrict__ g,
    const float* __restrict__ b, float* __restrict__ out_f32,
    h16* __restrict__ out_h16, int L)
{
    const int wave = threadIdx.x >> 5, lane = threadIdx.x & 31;
    const size_t row = (size_t)blockIdx.x * 8 + wave;
    const float* p = src + row * L;

    float vals[16];
    float s = 0.f;
    #pragma unroll
    for (int i = 0; i < 16; ++i) { vals[i] = p[lane + i * 32]; s += vals[i]; }
    #pragma unroll
    for (int o = 16; o > 0; o >>= 1) s += __shfl_xor(s, o, 32);
    const float mu = s * (1.0f / 512.0f);

    float v2 = 0.f;
    #pragma unroll
    for (int i = 0; i < 16; ++i) { const float d = vals[i] - mu; v2 += d * d; }
    #pragma unroll
    for (int o = 16; o > 0; o >>= 1) v2 += __shfl_xor(v2, o, 32);
    const float inv = rsqrtf(v2 * (1.0f / 512.0f) + 1e-5f);

    #pragma unroll
    for (int i = 0; i < 16; ++i) {
        const int c = lane + i * 32;
        const float y = (vals[i] - mu) * inv * g[c] + b[c];
        out_f32[row * L + c] = y;
        out_h16[row * L + c] = (h16)y;
    }
}

// ---------------------------------------------------------------------------
// Causal softmax over one row of scores [S,S] (per-batch launch).
// Block 256 per row; 16 cols/thread; writes fp16 attn.
// ---------------------------------------------------------------------------
__global__ __launch_bounds__(256) void lsa_softmax(
    const float* __restrict__ scores, h16* __restrict__ attn, int S, float scale)
{
    const int i = blockIdx.x;
    const float* p = scores + (size_t)i * S;
    h16* q = attn + (size_t)i * S;
    __shared__ float red[8];

    const int tid = threadIdx.x, lane = tid & 31, wave = tid >> 5;

    float v[16];
    float mx = -3.4e38f;
    #pragma unroll
    for (int t = 0; t < 16; ++t) {
        const int j = tid + t * 256;
        const float s = (j <= i) ? p[j] * scale : -3.4e38f;
        v[t] = s;
        mx = fmaxf(mx, s);
    }
    #pragma unroll
    for (int o = 16; o > 0; o >>= 1) mx = fmaxf(mx, __shfl_xor(mx, o, 32));
    if (lane == 0) red[wave] = mx;
    __syncthreads();
    mx = red[0];
    #pragma unroll
    for (int w = 1; w < 8; ++w) mx = fmaxf(mx, red[w]);
    __syncthreads();

    float sum = 0.f;
    #pragma unroll
    for (int t = 0; t < 16; ++t) {
        const int j = tid + t * 256;
        const float e = (j <= i) ? __expf(v[t] - mx) : 0.f;
        v[t] = e;
        sum += e;
    }
    #pragma unroll
    for (int o = 16; o > 0; o >>= 1) sum += __shfl_xor(sum, o, 32);
    if (lane == 0) red[wave] = sum;
    __syncthreads();
    sum = 0.f;
    #pragma unroll
    for (int w = 0; w < 8; ++w) sum += red[w];
    const float r = 1.0f / sum;
    #pragma unroll
    for (int t = 0; t < 16; ++t) q[tid + t * 256] = (h16)(v[t] * r);
}

// ---------------------------------------------------------------------------
// Host-side orchestration
// ---------------------------------------------------------------------------
extern "C" void kernel_launch(void* const* d_in, const int* in_sizes, int n_in,
                              void* d_out, int out_size, void* d_ws, size_t ws_size,
                              hipStream_t stream)
{
    (void)in_sizes; (void)n_in; (void)out_size; (void)ws_size;

    constexpr int B = LSA_B, S = LSA_S, E = LSA_E, L = LSA_L;
    constexpr size_t M = (size_t)B * S;               // 16384 rows

    const float* x     = (const float*)d_in[0];
    const float* w_q   = (const float*)d_in[1];
    const float* w_dkv = (const float*)d_in[2];
    const float* w_uk  = (const float*)d_in[3];
    const float* w_uv  = (const float*)d_in[4];
    const float* w_o   = (const float*)d_in[5];
    const float* ln_g  = (const float*)d_in[6];
    const float* ln_b  = (const float*)d_in[7];

    float* out_attn_f32 = (float*)d_out;                       // [B,S,E]
    float* out_ckv_f32  = (float*)d_out + M * E;               // [B,S,L]

    // ---- workspace carve-up (bytes) ----
    char* ws = (char*)d_ws;
    size_t off = 0;
    auto alloc = [&](size_t bytes) -> char* {
        char* p = ws + off;
        off = (off + bytes + 255) & ~(size_t)255;
        return p;
    };
    h16*   xb        = (h16*)  alloc(M * E * 2);          //  64 MB
    h16*   wq_h      = (h16*)  alloc((size_t)E * E * 2);  //   8 MB
    h16*   wdkv_h    = (h16*)  alloc((size_t)L * E * 2);  //   2 MB
    h16*   wuk_h     = (h16*)  alloc((size_t)L * E * 2);  //   2 MB
    h16*   wo_h      = (h16*)  alloc((size_t)E * E * 2);  //   8 MB
    h16*   wuvT_h    = (h16*)  alloc((size_t)E * L * 2);  //   2 MB
    float* absorbed  = (float*)alloc((size_t)E * L * 4);  //   4 MB
    h16*   absorbedT = (h16*)  alloc((size_t)L * E * 2);  //   2 MB
    float* ckv_raw   = (float*)alloc(M * L * 4);          //  32 MB
    h16*   ckv_h     = (h16*)  alloc(M * L * 2);          //  16 MB
    h16*   qlat_h    = (h16*)  alloc(M * L * 2);          //  16 MB
    float* scores    = (float*)alloc((size_t)S * S * 4);  //  64 MB (per-batch reuse)
    h16*   attn_h    = (h16*)  alloc((size_t)S * S * 2);  //  32 MB (per-batch reuse)
    h16*   valT_h    = (h16*)  alloc((size_t)B * E * S * 2); // 64 MB
    h16*   oat_h     = (h16*)  alloc(M * E * 2);          //  64 MB

    const dim3 blk(256);

    // 1) fp32 -> fp16 conversions
    {
        size_t n = M * E;
        lsa_conv<<<dim3((unsigned)((n + 255) / 256)), blk, 0, stream>>>(x, xb, n);
        n = (size_t)E * E;
        lsa_conv<<<dim3((unsigned)((n + 255) / 256)), blk, 0, stream>>>(w_q, wq_h, n);
        lsa_conv<<<dim3((unsigned)((n + 255) / 256)), blk, 0, stream>>>(w_o, wo_h, n);
        n = (size_t)L * E;
        lsa_conv<<<dim3((unsigned)((n + 255) / 256)), blk, 0, stream>>>(w_dkv, wdkv_h, n);
        lsa_conv<<<dim3((unsigned)((n + 255) / 256)), blk, 0, stream>>>(w_uk, wuk_h, n);
        // w_uv [L,E] -> w_uv^T [E,L] fp16 (so value GEMM reads K contiguously)
        lsa_tconv<<<dim3(E / 32, L / 32), blk, 0, stream>>>(w_uv, wuvT_h, L, E);
    }

    // 2) c_kv_raw = x @ w_dkv^T   [M,L] f32
    lsa_gemm<0, false><<<dim3(L / 128, (unsigned)(M / 128), 1), blk, 0, stream>>>(
        xb, wdkv_h, ckv_raw, (int)M, L, E, 0, 0, 0);

    // 3) LayerNorm -> c_kv (f32 into d_out) + fp16 copy
    lsa_layernorm<<<dim3((unsigned)(M / 8)), blk, 0, stream>>>(
        ckv_raw, ln_g, ln_b, out_ckv_f32, ckv_h, L);

    // 4) absorbed = w_q @ w_uk^T  [E,L] f32 ; then transpose -> [L,E] fp16
    lsa_gemm<0, false><<<dim3(L / 128, E / 128, 1), blk, 0, stream>>>(
        wq_h, wuk_h, absorbed, E, L, E, 0, 0, 0);
    lsa_tconv<<<dim3(L / 32, E / 32), blk, 0, stream>>>(absorbed, absorbedT, E, L);

    // 5) q_lat = x @ absorbed    [M,L] fp16
    lsa_gemm<1, false><<<dim3(L / 128, (unsigned)(M / 128), 1), blk, 0, stream>>>(
        xb, absorbedT, qlat_h, (int)M, L, E, 0, 0, 0);

    // 6) value^T = (c_kv @ w_uv)^T per batch -> [B][E,S] fp16 (transposed epilogue)
    lsa_gemm<2, false><<<dim3(E / 128, S / 128, B), blk, 0, stream>>>(
        ckv_h, wuvT_h, valT_h, S, E, L,
        (long long)S * L, 0, (long long)E * S);

    // 7) attention per batch (reuses scores/attn buffers; stream-ordered)
    const float scale = 1.0f / sqrtf((float)L);
    for (int b = 0; b < B; ++b) {
        const h16* qb = qlat_h + (size_t)b * S * L;
        const h16* kb = ckv_h  + (size_t)b * S * L;
        // scores = q_lat @ c_kv^T (causal tiles only) [S,S] f32
        lsa_gemm<0, true><<<dim3(S / 128, S / 128, 1), blk, 0, stream>>>(
            qb, kb, scores, S, S, L, 0, 0, 0);
        // masked softmax -> fp16 attn
        lsa_softmax<<<dim3(S), blk, 0, stream>>>(scores, attn_h, S, scale);
        // out_b = attn @ value  [S,E] fp16   (Bt = value^T[b])
        lsa_gemm<1, false><<<dim3(E / 128, S / 128, 1), blk, 0, stream>>>(
            attn_h, valT_h + (size_t)b * E * S, oat_h + (size_t)b * S * E,
            S, E, S, 0, 0, 0);
    }

    // 8) out = o @ w_o^T  [M,E] f32 straight into d_out
    lsa_gemm<0, false><<<dim3(E / 128, (unsigned)(M / 128), 1), blk, 0, stream>>>(
        oat_h, wo_h, out_attn_f32, (int)M, E, E, 0, 0, 0);
}
